// InteractionBlock_24412594111108
// MI455X (gfx1250) — matrix operand order repaired and checked by reading, not compiled, not run on hardware
//
#include <hip/hip_runtime.h>
#include <hip/hip_bf16.h>

// CDNA5 / gfx1250, wave32. WMMA f16->f32 for all matmuls, fp32 messages/scatter.
typedef __attribute__((ext_vector_type(16))) _Float16 v16h;
typedef __attribute__((ext_vector_type(8)))  float    v8f;
typedef __attribute__((ext_vector_type(4)))  unsigned int u32x4;
typedef __attribute__((ext_vector_type(8)))  int      i32x8;
typedef __attribute__((ext_vector_type(4)))  int      i32x4;

#define WMMA_F32_F16(A, B, C) \
  __builtin_amdgcn_wmma_f32_16x16x32_f16(false, (A), false, (B), (short)0, (C), false, false)

#if defined(__has_builtin)
#if __has_builtin(__builtin_amdgcn_tensor_load_to_lds)
#define HAVE_TDM 1
#endif
#endif

// ---------------------------------------------------------------------------
// Zero the (N,256) scatter accumulator (must be re-zeroed every launch).
// ---------------------------------------------------------------------------
__global__ void zero_kernel(float4* __restrict__ p, long n4) {
  long i = (long)blockIdx.x * blockDim.x + threadIdx.x;
  long stride = (long)gridDim.x * blockDim.x;
  float4 z = {0.f, 0.f, 0.f, 0.f};
  for (; i < n4; i += stride) p[i] = z;
}

// ---------------------------------------------------------------------------
// Node prep: h0 = alpha/sqrt(32) * x0 @ lin1_w0 ; h1_m = alpha/sqrt(32) * x1_m @ lin1_w1
// h layout: [h0(32) | h1 as (v*3+m)(96)]  -> (N,128)
// A layout (16-bit 16x32): lane L<16 holds row L, K in {0..7,16..23};
//                          lane L+16 holds row L, K in {8..15,24..31}.
// B layout (16-bit 32x16): element e of lane n: K = (e&7) + 16*(e>=8) + 8*(n>=16), N = n&15.
// ---------------------------------------------------------------------------
__global__ void __launch_bounds__(128)
node_kernel(const float* __restrict__ nf, const float* __restrict__ lin1_w0,
            const float* __restrict__ lin1_w1, float* __restrict__ h, int N) {
  __shared__ v16h sB0[2][32];   // lin1_w0 fragments, 2 col tiles
  __shared__ v16h sB1[2][32];   // lin1_w1 fragments
  const int tid = threadIdx.x;
  const float S1 = 0.25f * 0.17677669529663687f; // alpha(=1/4) / sqrt(32)
  {
    _Float16* p0 = (_Float16*)sB0;
    _Float16* p1 = (_Float16*)sB1;
    for (int f = tid; f < 2 * 32 * 16; f += blockDim.x) {
      int e = f & 15, lane = (f >> 4) & 31, t = f >> 9;
      int K = (e & 7) + ((e & 8) ? 16 : 0) + ((lane & 16) ? 8 : 0);
      int col = 16 * t + (lane & 15);
      p0[f] = (_Float16)(lin1_w0[K * 32 + col] * S1);
      p1[f] = (_Float16)(lin1_w1[K * 32 + col] * S1);
    }
  }
  __syncthreads();

  const int wave = tid >> 5, lane = tid & 31, laneLo = lane & 15;
  const bool hiHalf = lane >= 16;
  // force wave-uniform tile base into an SGPR -> scalar guard branches
  const int n_base = __builtin_amdgcn_readfirstlane(blockIdx.x * 64 + wave * 16);
  const bool fullTile = (n_base + 16 <= N);           // scalar compare
  const int rowA = (n_base + laneLo < N) ? (n_base + laneLo) : (N - 1);
  const float* xr = nf + (size_t)rowA * 128;

  // ---- h0 = x0 @ W0 ----
  v16h a0;
#pragma unroll
  for (int ee = 0; ee < 16; ++ee) {
    int K = ee + ((ee & 8) ? 8 : 0) + (hiHalf ? 8 : 0);
    a0[ee] = (_Float16)xr[K];
  }
#pragma unroll
  for (int t = 0; t < 2; ++t) {
    v8f c = {};
    c = WMMA_F32_F16(a0, sB0[t][lane], c);
    int col = 16 * t + laneLo;
    if (fullTile) {
#pragma unroll
      for (int r = 0; r < 8; ++r) {
        int node = n_base + (hiHalf ? 8 : 0) + r;
        h[(size_t)node * 128 + col] = c[r];
      }
    } else {
#pragma unroll
      for (int r = 0; r < 8; ++r) {
        int node = n_base + (hiHalf ? 8 : 0) + r;
        if (node < N) h[(size_t)node * 128 + col] = c[r];
      }
    }
  }
  // ---- h1_m = x1_m @ W1, m = 0..2 ----
#pragma unroll
  for (int m = 0; m < 3; ++m) {
    v16h a1;
#pragma unroll
    for (int ee = 0; ee < 16; ++ee) {
      int K = ee + ((ee & 8) ? 8 : 0) + (hiHalf ? 8 : 0);
      a1[ee] = (_Float16)xr[32 + K * 3 + m];
    }
#pragma unroll
    for (int t = 0; t < 2; ++t) {
      v8f c = {};
      c = WMMA_F32_F16(a1, sB1[t][lane], c);
      int col = 16 * t + laneLo;
      if (fullTile) {
#pragma unroll
        for (int r = 0; r < 8; ++r) {
          int node = n_base + (hiHalf ? 8 : 0) + r;
          h[(size_t)node * 128 + 32 + col * 3 + m] = c[r];
        }
      } else {
#pragma unroll
        for (int r = 0; r < 8; ++r) {
          int node = n_base + (hiHalf ? 8 : 0) + r;
          if (node < N) h[(size_t)node * 128 + 32 + col * 3 + m] = c[r];
        }
      }
    }
  }
}

// ---------------------------------------------------------------------------
// Edge kernel: per wave = 16 edges.
//   A-panel (16x8 f32 embedding tile) DMA'd to LDS by the Tensor Data Mover,
//   overlapped with weight-fragment staging.
//   W = silu(emb @ W0/sqrt8) @ W1/sqrt64   (16x8 -> 16x64 -> 16x128, WMMA)
//   then message products + fp32 atomic scatter into mn (N,256).
// ---------------------------------------------------------------------------
__global__ void __launch_bounds__(128)
edge_kernel(const float* __restrict__ edge_attrs, const float* __restrict__ edge_emb,
            const int* __restrict__ edge_index, const float* __restrict__ mlp_w0,
            const float* __restrict__ mlp_w1, const float* __restrict__ h,
            float* __restrict__ mn, int N, int E) {
  __shared__ v16h sW0frag[4][32];       // (8->32 K-padded) x 64, 4 col tiles
  __shared__ v16h sW1frag[8][2][32];    // 64 x 128, 8 col tiles x 2 K-chunks
  __shared__ _Float16 sAct[4][16][72];  // per-wave silu activations (16x64 + pad)
  __shared__ float sWmsg[4][16][128];   // per-wave edge weights W, fp32
  __shared__ float sEmb[4][16][8];      // per-wave embedding tile (TDM dest)

  const int tid = threadIdx.x;
  const int wave = tid >> 5, lane = tid & 31, laneLo = lane & 15;
  const bool hiHalf = lane >= 16;
  const int e_base = __builtin_amdgcn_readfirstlane(blockIdx.x * 64 + wave * 16);

#if HAVE_TDM
  // ---- issue TDM: 2D tile load, data_size=4B, tile 8(x) x 16(y), row stride 8.
  // tensor dims from the tile start: dim0=8, dim1=E-e_base (OOB rows read 0).
  {
    unsigned lds_off = (unsigned)(uintptr_t)&sEmb[wave][0][0]; // addr[31:0] = LDS offset
    unsigned long long ga =
        (unsigned long long)(uintptr_t)(edge_emb + (size_t)e_base * 8);
    unsigned rows = (unsigned)(E - e_base);
    u32x4 g0;
    g0[0] = 1u;                                  // count=1, user D#
    g0[1] = lds_off;                             // lds_addr (bytes)
    g0[2] = (unsigned)(ga & 0xffffffffu);        // global_addr[31:0]
    g0[3] = (unsigned)((ga >> 32) & 0x1ffffffu)  // global_addr[56:32]
            | (2u << 30);                        // type = 2 ("image")
    i32x8 g1;
    g1[0] = (int)(2u << 16);                     // data_size = 4 bytes
    g1[1] = (int)(8u << 16);                     // tensor_dim0 = 8 (bits 79:48 lo)
    g1[2] = (int)((rows & 0xffffu) << 16);       // tensor_dim1 lo16 (bits 111:80)
    g1[3] = (int)((rows >> 16) | (8u << 16));    // tensor_dim1 hi16 | tile_dim0=8
    g1[4] = 16;                                  // tile_dim1 = 16 rows
    g1[5] = 8;                                   // tensor_dim0_stride = 8
    g1[6] = 0;
    g1[7] = 0;
    i32x4 gz4 = {0, 0, 0, 0};                    // no dim2/3 (2D tensor)
    i32x8 gz8 = {0, 0, 0, 0, 0, 0, 0, 0};
    __builtin_amdgcn_tensor_load_to_lds(g0, g1, gz4, gz4, gz8, 0);
  }
#endif
  // prefetch the index/attr lines the message phase will touch
  {
    int e = e_base + (lane & 15);
    if (e < E) {
      __builtin_prefetch(edge_index + e, 0, 1);
      __builtin_prefetch(edge_index + E + e, 0, 1);
      __builtin_prefetch(edge_attrs + (size_t)e * 4, 0, 1);
    }
  }

  // ---- stage swizzled f16 weight fragments (overlaps the TDM) ----
  {
    _Float16* p = (_Float16*)sW0frag;
    for (int f = tid; f < 4 * 32 * 16; f += blockDim.x) {
      int e = f & 15, ln = (f >> 4) & 31, t = f >> 9;
      int K = e + ((ln & 16) ? 16 : 0);
      int col = 16 * t + (ln & 15);
      float v = (K < 8) ? mlp_w0[K * 64 + col] * 0.35355339f : 0.0f; // 1/sqrt(8)
      p[f] = (_Float16)v;
    }
    _Float16* q = (_Float16*)sW1frag;
    for (int f = tid; f < 8 * 2 * 32 * 16; f += blockDim.x) {
      int e = f & 15, ln = (f >> 4) & 31, k = (f >> 9) & 1, t = f >> 10;
      int K = 32 * k + (e & 7) + ((e & 8) ? 16 : 0) + ((ln & 16) ? 8 : 0);
      int col = 16 * t + (ln & 15);
      q[f] = (_Float16)(mlp_w1[K * 128 + col] * 0.125f); // 1/sqrt(64)
    }
  }
#if !HAVE_TDM
  {
    int erow = e_base + laneLo;
    if (!hiHalf && erow < E) {
      const float* em = edge_emb + (size_t)erow * 8;
#pragma unroll
      for (int i = 0; i < 8; ++i) sEmb[wave][laneLo][i] = em[i];
    }
  }
#endif
  __syncthreads();
#if HAVE_TDM
  __builtin_amdgcn_s_wait_tensorcnt(0);  // this wave's tile is in LDS
#endif

  // ---- layer 0 A: 16x8 embedding, K-padded to 32 (hi lanes hold K>=8 == 0) ----
  v16h a0 = {};
  if (!hiHalf) {
#pragma unroll
    for (int i = 0; i < 8; ++i) a0[i] = (_Float16)sEmb[wave][laneLo][i];
  }
  // ---- layer 0 WMMA + SiLU, stage activations (row-major f16) ----
#pragma unroll
  for (int t = 0; t < 4; ++t) {
    v8f c = {};
    c = WMMA_F32_F16(a0, sW0frag[t][lane], c);
#pragma unroll
    for (int r = 0; r < 8; ++r) {
      float x = c[r];
      float s = x / (1.0f + __expf(-x)); // silu
      int row = (hiHalf ? 8 : 0) + r;
      sAct[wave][row][16 * t + laneLo] = (_Float16)s;
    }
  }
  // ---- layer 1: (16x64) @ (64x128), 2 K-chunks x 8 N-tiles ----
#pragma unroll
  for (int t = 0; t < 8; ++t) {
    v8f acc = {};
#pragma unroll
    for (int k = 0; k < 2; ++k) {
      const _Float16* ap = &sAct[wave][laneLo][0];
      int kb = 32 * k + (hiHalf ? 8 : 0);
      v16h a;
#pragma unroll
      for (int i = 0; i < 8; ++i) {
        a[i]     = ap[kb + i];       // K = kb..kb+7
        a[8 + i] = ap[kb + 16 + i];  // K = kb+16..kb+23
      }
      acc = WMMA_F32_F16(a, sW1frag[t][k][lane], acc);
    }
#pragma unroll
    for (int r = 0; r < 8; ++r) {
      int row = (hiHalf ? 8 : 0) + r;
      sWmsg[wave][row][16 * t + laneLo] = acc[r];
    }
  }

  // ---- message phase: lane -> (edge j = lane/2, u half = 16*(lane&1)) ----
  // wave-private LDS; DS ops are in program order within a wave, no barrier needed.
  const int j = lane >> 1;
  const int uh = (lane & 1) * 16;
  const int e = e_base + j;
  const bool fullE = (e_base + 16 <= E);   // scalar fast path around the guard
  if (fullE || e < E) {
    int dst = edge_index[e];
    int src = edge_index[E + e];
    const float* ea = edge_attrs + (size_t)e * 4;
    float y0 = ea[0], y1x = ea[1], y1y = ea[2], y1z = ea[3];
    const float* hs = h + (size_t)src * 128;
    float* outp = mn + (size_t)dst * 256;
    const float* wrow = &sWmsg[wave][j][0];
#pragma unroll
    for (int i = 0; i < 16; ++i) {
      int u = uh + i;
      float wa = wrow[u], wb = wrow[32 + u], wc = wrow[64 + u], wd = wrow[96 + u];
      float x0  = hs[u];
      float x1x = hs[32 + u * 3 + 0];
      float x1y = hs[32 + u * 3 + 1];
      float x1z = hs[32 + u * 3 + 2];
      float dotp = x1x * y1x + x1y * y1y + x1z * y1z;
      atomicAdd(outp + u,                 wa * x0 * y0);
      atomicAdd(outp + 32 + u,            wd * dotp * 0.57735027f); // 1/sqrt(3)
      atomicAdd(outp + 64 + u * 3 + 0,    wb * x0 * y1x);
      atomicAdd(outp + 64 + u * 3 + 1,    wb * x0 * y1y);
      atomicAdd(outp + 64 + u * 3 + 2,    wb * x0 * y1z);
      atomicAdd(outp + 160 + u * 3 + 0,   wc * x1x * y0);
      atomicAdd(outp + 160 + u * 3 + 1,   wc * x1y * y0);
      atomicAdd(outp + 160 + u * 3 + 2,   wc * x1z * y0);
    }
  }
}

// ---------------------------------------------------------------------------
// Output: o0 = n0 @ lin2_w0/sqrt64 + sc0 ; o1_m = n1_m @ lin2_w1/sqrt64 + sc1_m
// sc uses the species-selected slice of sc_w* (per-row weight -> VALU).
// ---------------------------------------------------------------------------
__global__ void __launch_bounds__(128)
out_kernel(const float* __restrict__ mn, const float* __restrict__ nf,
           const float* __restrict__ node_attrs,
           const float* __restrict__ lin2_w0, const float* __restrict__ lin2_w1,
           const float* __restrict__ sc_w0, const float* __restrict__ sc_w1,
           float* __restrict__ out, int N) {
  __shared__ v16h sB0[2][2][32];  // lin2_w0: 2 col tiles x 2 K-chunks
  __shared__ v16h sB1[2][2][32];  // lin2_w1
  __shared__ int sSpec[4][16];
  const int tid = threadIdx.x;
  {
    _Float16* p0 = (_Float16*)sB0;
    _Float16* p1 = (_Float16*)sB1;
    for (int f = tid; f < 2 * 2 * 32 * 16; f += blockDim.x) {
      int e = f & 15, lane = (f >> 4) & 31, k = (f >> 9) & 1, t = f >> 10;
      int K = 32 * k + (e & 7) + ((e & 8) ? 16 : 0) + ((lane & 16) ? 8 : 0);
      int col = 16 * t + (lane & 15);
      p0[f] = (_Float16)(lin2_w0[K * 32 + col] * 0.125f); // 1/sqrt(64)
      p1[f] = (_Float16)(lin2_w1[K * 32 + col] * 0.125f);
    }
  }
  const int wave = tid >> 5, lane = tid & 31, laneLo = lane & 15;
  const bool hiHalf = lane >= 16;
  const int n_base = __builtin_amdgcn_readfirstlane(blockIdx.x * 64 + wave * 16);
  const bool fullTile = (n_base + 16 <= N);   // scalar compare
  if (!hiHalf) {  // species of the wave's 16 nodes (one-hot argmax)
    int node = (n_base + laneLo < N) ? (n_base + laneLo) : (N - 1);
    int s = 0;
    for (int v = 0; v < 16; ++v)
      if (node_attrs[(size_t)node * 16 + v] > 0.5f) s = v;
    sSpec[wave][laneLo] = s;
  }
  __syncthreads();

  const int rowA = (n_base + laneLo < N) ? (n_base + laneLo) : (N - 1);
  const float* mr = mn + (size_t)rowA * 256;
  const float FAN_SC = 0.044194173824159216f; // 1/sqrt(32*16)

  // ---- o0 ----
  {
    v16h ak0, ak1;
#pragma unroll
    for (int ee = 0; ee < 16; ++ee) {
      int Kb = ee + ((ee & 8) ? 8 : 0) + (hiHalf ? 8 : 0);
      ak0[ee] = (_Float16)mr[Kb];
      ak1[ee] = (_Float16)mr[32 + Kb];
    }
#pragma unroll
    for (int t = 0; t < 2; ++t) {
      v8f c = {};
      c = WMMA_F32_F16(ak0, sB0[t][0][lane], c);
      c = WMMA_F32_F16(ak1, sB0[t][1][lane], c);
      int col = 16 * t + laneLo;
#pragma unroll
      for (int r = 0; r < 8; ++r) {
        int M = (hiHalf ? 8 : 0) + r;
        int node = n_base + M;
        if (fullTile || node < N) {
          int s = sSpec[wave][M];
          const float* x0 = nf + (size_t)node * 128;
          float sc = 0.f;
          for (int u = 0; u < 32; ++u)
            sc += x0[u] * sc_w0[(size_t)u * 512 + s * 32 + col];
          out[(size_t)node * 128 + col] = c[r] + FAN_SC * sc;
        }
      }
    }
  }
  // ---- o1, m = 0..2 ----
#pragma unroll
  for (int m = 0; m < 3; ++m) {
    v16h ak0, ak1;
#pragma unroll
    for (int ee = 0; ee < 16; ++ee) {
      int Kb = ee + ((ee & 8) ? 8 : 0) + (hiHalf ? 8 : 0);
      ak0[ee] = (_Float16)mr[64 + Kb * 3 + m];
      ak1[ee] = (_Float16)mr[64 + (32 + Kb) * 3 + m];
    }
#pragma unroll
    for (int t = 0; t < 2; ++t) {
      v8f c = {};
      c = WMMA_F32_F16(ak0, sB1[t][0][lane], c);
      c = WMMA_F32_F16(ak1, sB1[t][1][lane], c);
      int col = 16 * t + laneLo;
#pragma unroll
      for (int r = 0; r < 8; ++r) {
        int M = (hiHalf ? 8 : 0) + r;
        int node = n_base + M;
        if (fullTile || node < N) {
          int s = sSpec[wave][M];
          const float* xr = nf + (size_t)node * 128;
          float sc = 0.f;
          for (int u = 0; u < 32; ++u)
            sc += xr[32 + u * 3 + m] * sc_w1[(size_t)u * 512 + s * 32 + col];
          out[(size_t)node * 128 + 32 + col * 3 + m] = c[r] + FAN_SC * sc;
        }
      }
    }
  }
}

// ---------------------------------------------------------------------------
extern "C" void kernel_launch(void* const* d_in, const int* in_sizes, int n_in,
                              void* d_out, int out_size, void* d_ws, size_t ws_size,
                              hipStream_t stream) {
  (void)n_in; (void)out_size; (void)ws_size;
  const float* node_features = (const float*)d_in[0];
  const float* node_attrs    = (const float*)d_in[1];
  const float* edge_attrs    = (const float*)d_in[2];
  const float* edge_emb      = (const float*)d_in[3];
  const int*   edge_index    = (const int*)d_in[4];
  const float* lin1_w0 = (const float*)d_in[5];
  const float* lin1_w1 = (const float*)d_in[6];
  const float* mlp_w0  = (const float*)d_in[7];
  const float* mlp_w1  = (const float*)d_in[8];
  const float* lin2_w0 = (const float*)d_in[9];
  const float* lin2_w1 = (const float*)d_in[10];
  const float* sc_w0   = (const float*)d_in[11];
  const float* sc_w1   = (const float*)d_in[12];
  float* out = (float*)d_out;

  const int N = in_sizes[0] / 128;
  const int E = in_sizes[4] / 2;

  // Workspace: h (N,128) f32 | mn (N,256) f32   (~61.4 MB; mn is L2-resident)
  char* ws = (char*)d_ws;
  float* h = (float*)ws;
  size_t h_bytes = (((size_t)N * 128 * sizeof(float)) + 255) & ~(size_t)255;
  float* mn = (float*)(ws + h_bytes);

  long n4 = ((long)N * 256) / 4;
  zero_kernel<<<1024, 256, 0, stream>>>((float4*)mn, n4);

  int nb_node = (N + 63) / 64;
  node_kernel<<<nb_node, 128, 0, stream>>>(node_features, lin1_w0, lin1_w1, h, N);

  int nb_edge = (E + 63) / 64;
  edge_kernel<<<nb_edge, 128, 0, stream>>>(edge_attrs, edge_emb, edge_index,
                                           mlp_w0, mlp_w1, h, mn, N, E);

  out_kernel<<<nb_node, 128, 0, stream>>>(mn, node_features, node_attrs,
                                          lin2_w0, lin2_w1, sc_w0, sc_w1, out, N);
}
